// Network_23527830847570
// MI455X (gfx1250) — matrix operand order repaired and checked by reading, not compiled, not run on hardware
//
#include <hip/hip_runtime.h>

typedef __attribute__((ext_vector_type(16))) _Float16 v16h;
typedef __attribute__((ext_vector_type(8)))  float    v8f;

// ---------------- problem constants ----------------
#define TT    15
#define CIN   2
#define HIN   192
#define C1    32
#define PH    97            // pooled H/W
#define HP    99            // padded pooled H/W
#define C2    150
#define H2    98
#define NPIX  (H2*H2)       // 9604
#define KDIM  128           // 32*2*2
#define MT    10            // ceil(150/16)
#define KS    4             // 128/32
#define NT    ((NPIX+15)/16) // 601

// ---------------- stage 1: conv1 + fire + maxpool ----------------
__global__ void k_conv1_pool(const float* __restrict__ x, const float* __restrict__ w1,
                             float* __restrict__ spk_pool) {
    __shared__ float wsh[CIN*5*5];
    const int c = blockIdx.y, t = blockIdx.z;
    if (threadIdx.x < CIN*5*5) wsh[threadIdx.x] = w1[c*CIN*5*5 + threadIdx.x];
    __syncthreads();
    const int pix = blockIdx.x*blockDim.x + threadIdx.x;
    if (pix >= PH*PH) return;
    const int ph = pix / PH, pw = pix % PH;
    const float* xt = x + (size_t)t*CIN*HIN*HIN;
    float best = 0.f;
    for (int dy = 0; dy < 2; ++dy) {
        const int r = ph*2 - 1 + dy;
        if (r < 0 || r >= HIN) continue;
        for (int dx = 0; dx < 2; ++dx) {
            const int cc = pw*2 - 1 + dx;
            if (cc < 0 || cc >= HIN) continue;
            float pot = 0.f;
            for (int ci = 0; ci < CIN; ++ci)
                for (int kh = 0; kh < 5; ++kh) {
                    const int y = r + kh - 2;
                    if ((unsigned)y >= (unsigned)HIN) continue;
                    for (int kw = 0; kw < 5; ++kw) {
                        const int xc = cc + kw - 2;
                        if ((unsigned)xc >= (unsigned)HIN) continue;
                        pot += xt[((size_t)ci*HIN + y)*HIN + xc] * wsh[(ci*5 + kh)*5 + kw];
                    }
                }
            best = fmaxf(best, (pot > 10.0f) ? 1.f : 0.f);
        }
    }
    spk_pool[(((size_t)t*C1 + c)*PH + ph)*PH + pw] = best;
}

// ---------------- stage 2: pointwise inhibition #1, emit f16 padded spikes ----------------
__global__ void k_inhibit1(const float* __restrict__ spk_pool, _Float16* __restrict__ spk_in) {
    const int idx = blockIdx.x*blockDim.x + threadIdx.x;
    if (idx >= HP*HP) return;
    const int h = idx / HP, w = idx % HP;
    if (h == 0 || h == HP-1 || w == 0 || w == HP-1) {
        for (int t = 0; t < TT; ++t)
            for (int c = 0; c < C1; ++c)
                spk_in[(((size_t)t*C1 + c)*HP + h)*HP + w] = (_Float16)0.f;
        return;
    }
    const int hh = h - 1, ww = w - 1;
    int count = 0; bool last = false;
    for (int t = 0; t < TT; ++t) {
        float mv = 0.f;
        for (int c = 0; c < C1; ++c)
            mv = fmaxf(mv, spk_pool[(((size_t)t*C1 + c)*PH + hh)*PH + ww]);
        const bool any = mv > 0.f;
        count += any ? 1 : 0;
        if (t == TT-1) last = any;
    }
    int earliest = TT - count;
    if (earliest < 0) earliest = 0;
    if (earliest > TT-1) earliest = TT-1;
    float mv = spk_pool[(((size_t)earliest*C1 + 0)*PH + hh)*PH + ww];
    int winner = 0;
    for (int c = 1; c < C1; ++c) {
        const float v = spk_pool[(((size_t)earliest*C1 + c)*PH + hh)*PH + ww];
        if (v > mv) { mv = v; winner = c; }
    }
    for (int t = 0; t < TT; ++t)
        for (int c = 0; c < C1; ++c) {
            float v = 0.f;
            if (last && c == winner)
                v = spk_pool[(((size_t)t*C1 + c)*PH + hh)*PH + ww];
            spk_in[(((size_t)t*C1 + c)*HP + h)*HP + w] = (_Float16)v;
        }
}

// ---------------- stage 3a: swizzle w2 into WMMA A-fragment layout (f16) ----------------
// A fragment (16-bit A 16x32): lane<16 -> M=lane,   halves j: K = j (j<8) else j+8
//                              lane>=16 -> M=lane-16, halves j: K = j+8 (j<8) else j+16
__global__ void k_prep_w2(const float* __restrict__ w2, _Float16* __restrict__ wA) {
    const int idx = blockIdx.x*blockDim.x + threadIdx.x;
    if (idx >= MT*KS*32*16) return;
    const int j = idx & 15, lane = (idx >> 4) & 31, ks = (idx >> 9) & 3, mt = idx >> 11;
    const int M = mt*16 + (lane & 15);
    int K;
    if (lane < 16) K = 32*ks + ((j < 8) ? j      : j + 8);
    else           K = 32*ks + ((j < 8) ? j + 8  : j + 16);
    // w2 flat (oc,ic,kh,kw): K == ic*4 + kh*2 + kw
    const float v = (M < C2) ? w2[(size_t)M*KDIM + K] : 0.f;
    wA[idx] = (_Float16)v;    // layout ((mt*KS+ks)*32+lane)*16+j == idx
}

// ---------------- stage 3b: conv2 implicit GEMM via WMMA (LDS-staged), fused fire ----------------
__global__ void __launch_bounds__(128) k_conv2_wmma(const _Float16* __restrict__ spk_in,
                                                    const _Float16* __restrict__ wA,
                                                    float* __restrict__ pot /* d_out */) {
    __shared__ __align__(32) _Float16 sA[MT*KS*32*16];   // 40960 B: all A fragments
    __shared__ __align__(32) _Float16 sB[C1*3*HP];       // 19008 B: 3 spike rows, all ic

    const int t    = blockIdx.y;
    const int tid  = threadIdx.x;
    const int wave = tid >> 5;
    const int lane = tid & 31;
    const int nt0  = blockIdx.x*4;       // first n-tile handled by this block
    const int p0   = nt0*16;             // < NPIX by grid construction
    const int oh0  = p0 / H2;
    const _Float16* sp = spk_in + (size_t)t*C1*HP*HP;

    // stage A fragments (coalesced dword copy of the pre-swizzled buffer)
    {
        const uint32_t* src = (const uint32_t*)wA;
        uint32_t*       dst = (uint32_t*)sA;
        for (int i = tid; i < MT*KS*32*16/2; i += 128) dst[i] = src[i];
    }
    // stage spike rows oh0..oh0+2, all 32 channels, full 99-wide rows
    for (int q = tid; q < C1*3*HP; q += 128) {
        const int ic  = q / (3*HP);
        const int rr  = (q / HP) % 3;
        const int col = q % HP;
        const int r   = oh0 + rr;
        _Float16 v = (_Float16)0.f;
        if (r < HP) v = sp[((size_t)ic*HP + r)*HP + col];
        sB[q] = v;
    }
    __syncthreads();

    const int ntile = nt0 + wave;        // wave-uniform
    if (ntile >= NT) return;             // only trailing waves of last block
    const int p = ntile*16 + (lane & 15);
    const bool pvalid = p < NPIX;
    // Clamp to a valid pixel for invalid lanes: their B column only feeds
    // output column p, which the epilogue never stores. Avoids predicated
    // per-element LDS gathers (no exec-mask churn, batched dscnt waits).
    const int pc = pvalid ? p : (NPIX - 1);
    const int dh = pc / H2 - oh0;        // 0..1 (clamped pixel stays in this block's rows)
    const int ow = pc % H2;

    v8f acc[MT] = {};
    for (int ks = 0; ks < KS; ++ks) {
        // B fragment (16-bit B 32x16): lane<16 holds K=0..7,16..23, lane>=16 K=8..15,24..31 (rel. ks*32), N=lane%16
        union { v16h v; _Float16 h[16]; } b;
        const int kb = ks*32 + ((lane < 16) ? 0 : 8);
#pragma unroll
        for (int j = 0; j < 16; ++j) {
            const int k  = kb + ((j < 8) ? j : j + 8);
            const int ic = k >> 2, kh = (k >> 1) & 1, kw = k & 1;
            b.h[j] = sB[(ic*3 + dh + kh)*HP + ow + kw];
        }
#pragma unroll
        for (int m = 0; m < MT; ++m) {
            const v16h a = *(const v16h*)(sA + ((m*KS + ks)*32 + lane)*16);
            acc[m] = __builtin_amdgcn_wmma_f32_16x16x32_f16(
                         false, a, false, b.v, (short)0, acc[m], false, false);
        }
    }
    // epilogue: C/D layout VGPR r -> M = r + (lane<16?0:8), N = lane%16; fire at thr=1
    const size_t outT = (size_t)t*C2*NPIX;
#pragma unroll
    for (int m = 0; m < MT; ++m) {
#pragma unroll
        for (int r = 0; r < 8; ++r) {
            const int oc = m*16 + r + ((lane < 16) ? 0 : 8);
            if (pvalid && oc < C2) {
                const float v = acc[m][r];
                pot[outT + (size_t)oc*NPIX + p] = (v > 1.0f) ? v : 0.f;
            }
        }
    }
}

// ---------------- stage 4: pointwise inhibition #2, in place on pot ----------------
__global__ void k_inhibit2(float* __restrict__ pot) {
    const int idx = blockIdx.x*blockDim.x + threadIdx.x;
    if (idx >= NPIX) return;
    int count = 0; bool last = false;
    for (int t = 0; t < TT; ++t) {
        const float* pt = pot + (size_t)t*C2*NPIX + idx;
        float mv = pt[0];
        for (int c = 1; c < C2; ++c) mv = fmaxf(mv, pt[(size_t)c*NPIX]);
        const bool any = mv > 0.f;
        count += any ? 1 : 0;
        if (t == TT-1) last = any;
    }
    int earliest = TT - count;
    if (earliest < 0) earliest = 0;
    if (earliest > TT-1) earliest = TT-1;
    const float* pe = pot + (size_t)earliest*C2*NPIX + idx;
    float mv = pe[0]; int winner = 0;
    for (int c = 1; c < C2; ++c) {
        const float v = pe[(size_t)c*NPIX];
        if (v > mv) { mv = v; winner = c; }
    }
    for (int t = 0; t < TT; ++t) {
        float* pt = pot + (size_t)t*C2*NPIX + idx;
        for (int c = 0; c < C2; ++c) {
            if (last && c == winner) continue;   // coef = 1: keep
            pt[(size_t)c*NPIX] = 0.f;
        }
    }
}

// ---------------- stage 5: k-winners ----------------
__global__ void k_zero(int* p) { if (threadIdx.x == 0 && blockIdx.x == 0) *p = 0; }

__global__ void k_kw_prepare(const float* __restrict__ pot, float* __restrict__ val,
                             float* __restrict__ nspk, int* __restrict__ vmax) {
    const int idx = blockIdx.x*blockDim.x + threadIdx.x;
    if (idx >= C2*NPIX) return;
    int nz = 0;
    for (int t = 0; t < TT; ++t)
        nz += (pot[(size_t)t*C2*NPIX + idx] > 0.f) ? 1 : 0;
    int earliest = TT - nz;
    if (earliest < 0) earliest = 0;
    if (earliest > TT-1) earliest = TT-1;
    const float v = pot[(size_t)earliest*C2*NPIX + idx];
    val[idx]  = v;
    nspk[idx] = (float)nz;
    const float tr = (nz > 0) ? v : 0.f;     // pot >= 0, so int-compare == float-compare
    atomicMax(vmax, __float_as_int(tr));
}

__global__ void k_kw_total(const float* __restrict__ val, const float* __restrict__ nspk,
                           const int* __restrict__ vmax, float* __restrict__ total) {
    const int idx = blockIdx.x*blockDim.x + threadIdx.x;
    if (idx >= C2*NPIX) return;
    const float v = __int_as_float(*vmax) * (float)TT;
    total[idx] = nspk[idx] * (val[idx] + v);
}

__global__ void __launch_bounds__(512) k_kw_select(float* __restrict__ total,
                                                   int* __restrict__ winners) {
    __shared__ float sv[512];
    __shared__ int   si[512];
    __shared__ int   wC, wH, wW, wValid;
    const int tid = threadIdx.x;
    const int N = C2*NPIX;
    for (int k = 0; k < 8; ++k) {
        float bv = -1e30f; int bi = 0x7fffffff;
        for (int i = tid; i < N; i += 512) {
            const float v = total[i];
            if (v > bv) { bv = v; bi = i; }    // strided scan keeps first (smallest) index
        }
        sv[tid] = bv; si[tid] = bi;
        __syncthreads();
        for (int s = 256; s > 0; s >>= 1) {
            if (tid < s) {
                const float v2 = sv[tid+s]; const int i2 = si[tid+s];
                if (v2 > sv[tid] || (v2 == sv[tid] && i2 < si[tid])) { sv[tid] = v2; si[tid] = i2; }
            }
            __syncthreads();
        }
        if (tid == 0) {
            const float mval = sv[0]; const int idx = si[0];
            const int valid = (mval != 0.f) ? 1 : 0;
            const int c = idx / NPIX, rem = idx % NPIX, h = rem / H2, w = rem % H2;
            winners[k*3+0] = valid ? c : -1;
            winners[k*3+1] = valid ? h : -1;
            winners[k*3+2] = valid ? w : -1;
            wC = c; wH = h; wW = w; wValid = valid;
        }
        __syncthreads();
        if (wValid) {
            const int c0 = wC, h0 = wH, w0 = wW;
            for (int i = tid; i < N; i += 512) {
                const int c = i / NPIX, rem = i % NPIX, h = rem / H2, w = rem % H2;
                const bool mask = (c == c0) || ((abs(h - h0) <= 1) && (abs(w - w0) <= 1));
                if (mask) total[i] = 0.f;
            }
        }
        __syncthreads();
    }
}

// ---------------- workspace layout ----------------
static constexpr size_t algn(size_t x) { return (x + 255) & ~size_t(255); }
static constexpr size_t OFF_POOL  = 0;
static constexpr size_t SZ_POOL   = (size_t)TT*C1*PH*PH*4;
static constexpr size_t OFF_SPKIN = algn(OFF_POOL + SZ_POOL);
static constexpr size_t SZ_SPKIN  = (size_t)TT*C1*HP*HP*2;
static constexpr size_t OFF_WA    = algn(OFF_SPKIN + SZ_SPKIN);
static constexpr size_t SZ_WA     = (size_t)MT*KS*32*16*2;
static constexpr size_t OFF_VAL   = algn(OFF_WA + SZ_WA);
static constexpr size_t SZ_CHW    = (size_t)C2*NPIX*4;
static constexpr size_t OFF_NSPK  = algn(OFF_VAL + SZ_CHW);
static constexpr size_t OFF_TOTAL = algn(OFF_NSPK + SZ_CHW);
static constexpr size_t OFF_VMAX  = algn(OFF_TOTAL + SZ_CHW);

extern "C" void kernel_launch(void* const* d_in, const int* in_sizes, int n_in,
                              void* d_out, int out_size, void* d_ws, size_t ws_size,
                              hipStream_t stream) {
    const float* x  = (const float*)d_in[0];   // (15,2,192,192)
    const float* w1 = (const float*)d_in[1];   // (32,2,5,5)
    const float* w2 = (const float*)d_in[2];   // (150,32,2,2)
    float* pot      = (float*)d_out;                              // (15,150,98,98)
    int*   winners  = (int*)d_out + (size_t)TT*C2*NPIX;           // (8,3) int32 bits

    char* ws = (char*)d_ws;
    float*    spk_pool = (float*)   (ws + OFF_POOL);
    _Float16* spk_in   = (_Float16*)(ws + OFF_SPKIN);
    _Float16* wA       = (_Float16*)(ws + OFF_WA);
    float*    val      = (float*)   (ws + OFF_VAL);
    float*    nspk     = (float*)   (ws + OFF_NSPK);
    float*    total    = (float*)   (ws + OFF_TOTAL);
    int*      vmax     = (int*)     (ws + OFF_VMAX);

    k_zero<<<1, 32, 0, stream>>>(vmax);

    dim3 g1((PH*PH + 255)/256, C1, TT);
    k_conv1_pool<<<g1, 256, 0, stream>>>(x, w1, spk_pool);

    k_inhibit1<<<(HP*HP + 255)/256, 256, 0, stream>>>(spk_pool, spk_in);

    k_prep_w2<<<(MT*KS*32*16 + 255)/256, 256, 0, stream>>>(w2, wA);

    dim3 g2((NT + 3)/4, TT);
    k_conv2_wmma<<<g2, 128, 0, stream>>>(spk_in, wA, pot);

    k_inhibit2<<<(NPIX + 255)/256, 256, 0, stream>>>(pot);

    k_kw_prepare<<<(C2*NPIX + 255)/256, 256, 0, stream>>>(pot, val, nspk, vmax);
    k_kw_total  <<<(C2*NPIX + 255)/256, 256, 0, stream>>>(val, nspk, vmax, total);
    k_kw_select <<<1, 512, 0, stream>>>(total, winners);
}
